// HybridTwoTower_65506841198664
// MI455X (gfx1250) — compile-verified, matrix operand order checked by reading, not run on hardware
//
#include <hip/hip_runtime.h>

#define N_ITEMS 100000
#define N_USERS 16384
#define HIST 50
#define TEXT_DIM 256
#define GCN_DIM 128
#define HIDDEN_DIM 512
#define TOWER_DIM 128

typedef __attribute__((ext_vector_type(16))) __bf16 v16bf;
typedef __attribute__((ext_vector_type(8)))  float  v8f;
typedef __attribute__((ext_vector_type(4)))  unsigned int v4u;
typedef __attribute__((ext_vector_type(2)))  unsigned int v2u;
typedef __attribute__((ext_vector_type(4)))  float  v4f;
typedef __attribute__((ext_vector_type(4)))  unsigned int u32x4;
typedef __attribute__((ext_vector_type(8)))  int i32x8;
typedef __attribute__((ext_vector_type(4)))  int i32x4;

union BFrag {
    v16bf v;
    v4u   q[2];          // 32 bytes = 16 bf16
    unsigned short s[16];
};

__device__ __forceinline__ unsigned short f2bf(float f) {
    unsigned int u = __float_as_uint(f);
    u += 0x7FFFu + ((u >> 16) & 1u);      // round-to-nearest-even
    return (unsigned short)(u >> 16);
}

__device__ __forceinline__ v8f wmma_bf16(v16bf a, v16bf b, v8f c) {
    return __builtin_amdgcn_wmma_f32_16x16x32_bf16(
        false, a, false, b, (short)0, c, false, false);
}

// ---------------------------------------------------------------------------
// f32 [K,N] -> bf16 fragment-swizzled: dst[((kt*NT+nt)*32+lane)*16+e]
// ---------------------------------------------------------------------------
__global__ void convert_weights_kernel(const float* __restrict__ src,
                                       unsigned short* __restrict__ dst,
                                       int K, int N) {
    int tid = blockIdx.x * blockDim.x + threadIdx.x;
    int NT = N >> 4;
    int KT = K >> 5;
    int total = KT * NT * 32;
    if (tid >= total) return;
    int lane = tid & 31;
    int tile = tid >> 5;
    int nt = tile % NT;
    int kt = tile / NT;
    int n = nt * 16 + (lane & 15);
    int kbase = kt * 32 + (lane >> 4) * 8;
    unsigned short out[16];
#pragma unroll
    for (int e = 0; e < 16; ++e) {
        int k = kbase + (e < 8 ? e : e + 8);
        out[e] = f2bf(src[(size_t)k * N + n]);
    }
    v4u* dp = (v4u*)&dst[(size_t)tid * 16];
    dp[0] = *(v4u*)&out[0];
    dp[1] = *(v4u*)&out[8];
}

// ---------------------------------------------------------------------------
// Item tower, M-tile = 64 (B fragments register-resident, swept over 4 M-subtiles)
// dynamic LDS: Xs[64][392] bf16 | Hs[64][520] bf16 | bias/red/scale
// Outs[64][132] f32 overlays Xs after GEMM1.
// ---------------------------------------------------------------------------
#define ITEM_XS_S 392
#define HS_S 520
#define OUT_S 132
#define ITEM_HS_OFF 50176                 // 64*392*2
#define ITEM_AUX_OFF (ITEM_HS_OFF + 66560)
#define ITEM_SMEM (ITEM_AUX_OFF + 2048 + 512 + 1024 + 256)   // 120576

__global__ __launch_bounds__(256) void item_tower_kernel(
    const float* __restrict__ text,
    const float* __restrict__ gcn,
    const unsigned short* __restrict__ w1,   // [12][32][32][16]
    const float* __restrict__ b1,
    const unsigned short* __restrict__ w2,   // [16][8][32][16]
    const float* __restrict__ b2,
    float* __restrict__ item_emb)
{
    extern __shared__ char smem[];
    unsigned short* Xs = (unsigned short*)smem;
    float*          Outs = (float*)smem;                       // overlay
    unsigned short* Hs = (unsigned short*)(smem + ITEM_HS_OFF);
    float* bias1 = (float*)(smem + ITEM_AUX_OFF);
    float* bias2 = bias1 + 512;
    float* red   = bias2 + 128;
    float* rowscale = red + 256;

    const int tid = threadIdx.x;
    const int w = tid >> 5, lane = tid & 31;
    const int rid = lane & 15, khalf = lane >> 4;
    const int base_row = blockIdx.x * 64;

    for (int i = tid; i < 512; i += 256) bias1[i] = b1[i];
    for (int i = tid; i < 128; i += 256) bias2[i] = b2[i];

    // stage X (convert f32 -> bf16): 64 x 384
    for (int i = 0; i < 24; ++i) {
        int f = tid + 256 * i;               // 0..6143
        int r = f / 96;
        int k = (f - r * 96) * 4;
        int gr = base_row + r; if (gr >= N_ITEMS) gr = N_ITEMS - 1;
        v4f d = (k < TEXT_DIM)
            ? *(const v4f*)&text[(size_t)gr * TEXT_DIM + k]
            : *(const v4f*)&gcn[(size_t)gr * GCN_DIM + (k - TEXT_DIM)];
        unsigned short o[4] = { f2bf(d.x), f2bf(d.y), f2bf(d.z), f2bf(d.w) };
        *(v2u*)&Xs[r * ITEM_XS_S + k] = *(v2u*)o;
    }
    __syncthreads();

    // ---- GEMM1: nt = w + 8j ; B panel in registers, swept over 4 mt -------
    for (int j = 0; j < 4; ++j) {
        int nt = w + 8 * j;
        if (j < 3)
            __builtin_prefetch(&w1[(((size_t)(w + 8 * (j + 1))) * 32 + lane) * 16], 0, 1);
        BFrag bf[12];
#pragma unroll
        for (int kt = 0; kt < 12; ++kt) {
            const v4u* bp = (const v4u*)&w1[(((size_t)kt * 32 + nt) * 32 + lane) * 16];
            bf[kt].q[0] = bp[0];
            bf[kt].q[1] = bp[1];
        }
        for (int mt = 0; mt < 4; ++mt) {
            const int arow = (mt * 16 + rid) * ITEM_XS_S;
            v8f c = {};
#pragma unroll
            for (int kt = 0; kt < 12; ++kt) {
                int k0 = kt * 32 + khalf * 8;
                BFrag af;
                af.q[0] = *(const v4u*)&Xs[arow + k0];
                af.q[1] = *(const v4u*)&Xs[arow + k0 + 16];
                c = wmma_bf16(af.v, bf[kt].v, c);
            }
            int col = nt * 16 + rid;
            float bv = bias1[col];
            int r0 = mt * 16 + khalf * 8;
#pragma unroll
            for (int v = 0; v < 8; ++v) {
                float x = c[v] + bv;
                x = x > 0.f ? x : 0.f;
                Hs[(r0 + v) * HS_S + col] = f2bf(x);
            }
        }
    }
    __syncthreads();

    // ---- GEMM2: wave w owns nt = w ; W2 read once per block ---------------
    {
        int nt = w;
        BFrag bf2[16];
#pragma unroll
        for (int kt = 0; kt < 16; ++kt) {
            const v4u* bp = (const v4u*)&w2[(((size_t)kt * 8 + nt) * 32 + lane) * 16];
            bf2[kt].q[0] = bp[0];
            bf2[kt].q[1] = bp[1];
        }
        for (int mt = 0; mt < 4; ++mt) {
            const int arow = (mt * 16 + rid) * HS_S;
            v8f c = {};
#pragma unroll
            for (int kt = 0; kt < 16; ++kt) {
                int k0 = kt * 32 + khalf * 8;
                BFrag af;
                af.q[0] = *(const v4u*)&Hs[arow + k0];
                af.q[1] = *(const v4u*)&Hs[arow + k0 + 16];
                c = wmma_bf16(af.v, bf2[kt].v, c);
            }
            int col = nt * 16 + rid;
            float bv = bias2[col];
            int r0 = mt * 16 + khalf * 8;
#pragma unroll
            for (int v = 0; v < 8; ++v)
                Outs[(r0 + v) * OUT_S + col] = c[v] + bv;   // overlays dead Xs
        }
    }
    __syncthreads();

    // ---- L2 norm: 4 threads/row, 32 elems each ----------------------------
    int r = tid >> 2, part = tid & 3;
    float s = 0.f;
#pragma unroll
    for (int i = 0; i < 32; ++i) {
        float x = Outs[r * OUT_S + part * 32 + i];
        s += x * x;
    }
    red[tid] = s;
    __syncthreads();
    if (part == 0) {
        float t = red[tid] + red[tid + 1] + red[tid + 2] + red[tid + 3];
        rowscale[r] = 1.f / fmaxf(sqrtf(t), 1e-12f);
    }
    __syncthreads();
    int gr = base_row + r;
    if (gr < N_ITEMS) {
        float sc = rowscale[r];
#pragma unroll
        for (int i = 0; i < 32; ++i)
            item_emb[(size_t)gr * TOWER_DIM + part * 32 + i] =
                Outs[r * OUT_S + part * 32 + i] * sc;
    }
}

// ---------------------------------------------------------------------------
// Pooling: one wave per user
// ---------------------------------------------------------------------------
__global__ __launch_bounds__(256) void pool_kernel(
    const float* __restrict__ item_emb,
    const int* __restrict__ hist_idx,
    const float* __restrict__ gcn_user,
    unsigned short* __restrict__ fused)
{
    int u = blockIdx.x * 8 + (threadIdx.x >> 5);
    int lane = threadIdx.x & 31;
    int d = lane * 4;
    float ax = 0.f, ay = 0.f, az = 0.f, aw = 0.f;
    const int* hp = &hist_idx[(size_t)u * HIST];
    for (int h = 0; h < HIST; ++h) {
        int idx = hp[h];
        v4f v = *(const v4f*)&item_emb[(size_t)idx * TOWER_DIM + d];
        ax += v.x; ay += v.y; az += v.z; aw += v.w;
    }
    const float inv = 1.0f / (float)HIST;
    v4f g = *(const v4f*)&gcn_user[(size_t)u * GCN_DIM + d];
    unsigned short go[4] = { f2bf(g.x), f2bf(g.y), f2bf(g.z), f2bf(g.w) };
    unsigned short ho[4] = { f2bf(ax * inv), f2bf(ay * inv),
                             f2bf(az * inv), f2bf(aw * inv) };
    *(v2u*)&fused[(size_t)u * 256 + d]       = *(v2u*)go;
    *(v2u*)&fused[(size_t)u * 256 + 128 + d] = *(v2u*)ho;
}

// ---------------------------------------------------------------------------
// User tower, M-tile = 64. A tile staged into LDS by the Tensor Data Mover:
// 2D D# tile (256 bf16 x 64 rows) with HW padding 4 DWORDs per 128 DWORDs,
// giving LDS row stride 528B = 264 bf16 (bank-conflict-free fragment reads).
// ---------------------------------------------------------------------------
#define USER_XS_S 264
#define USER_HS_OFF 33792                 // 64*264*2
#define USER_AUX_OFF (USER_HS_OFF + 66560)
#define USER_SMEM (USER_AUX_OFF + 2048 + 512 + 1024 + 256)   // 104192

__global__ __launch_bounds__(256) void user_tower_kernel(
    const unsigned short* __restrict__ fused, // [N_USERS,256] bf16
    const unsigned short* __restrict__ w1,    // [8][32][32][16]
    const float* __restrict__ b1,
    const unsigned short* __restrict__ w2,    // [16][8][32][16]
    const float* __restrict__ b2,
    float* __restrict__ out)
{
    extern __shared__ char smem[];
    unsigned short* Xs = (unsigned short*)smem;
    float*          Outs = (float*)smem;                       // overlay
    unsigned short* Hs = (unsigned short*)(smem + USER_HS_OFF);
    float* bias1 = (float*)(smem + USER_AUX_OFF);
    float* bias2 = bias1 + 512;
    float* red   = bias2 + 128;
    float* rowscale = red + 256;

    const int tid = threadIdx.x;
    const int w = tid >> 5, lane = tid & 31;
    const int rid = lane & 15, khalf = lane >> 4;
    const int base_row = blockIdx.x * 64;

    // ---- TDM: stage 64x256 bf16 tile -> LDS with HW padding ---------------
    if (w == 0) {
        unsigned long long gaddr =
            (unsigned long long)(const void*)&fused[(size_t)base_row * 256];
        unsigned int lds_off = (unsigned int)(size_t)(void*)Xs;
        u32x4 g0;
        i32x8 g1;
        i32x4 g2 = {0, 0, 0, 0};
        i32x4 g3 = {0, 0, 0, 0};
        i32x8 g4 = {0, 0, 0, 0, 0, 0, 0, 0};
        g0[0] = 1u;                                     // count = 1 valid D#
        g0[1] = lds_off;                                // lds_addr
        g0[2] = (unsigned int)(gaddr & 0xFFFFFFFFu);    // global_addr[31:0]
        g0[3] = (unsigned int)((gaddr >> 32) & 0x1FFFFFFu) | 0x80000000u; // [56:32] | type=2
        g1[0] = (int)0x07910000u;  // data_size=2B, pad_enable, pad_interval=6(128dw), pad_amount=3(4dw)
        g1[1] = (int)(256u << 16); // tensor_dim0 = 256 elements
        g1[2] = (int)((unsigned)N_USERS << 16);  // tensor_dim1 = N_USERS
        g1[3] = (int)(256u << 16); // tile_dim0 = 256
        g1[4] = 64;                // tile_dim1 = 64 rows
        g1[5] = 256;               // tensor_dim0_stride = 256 elements
        g1[6] = 0;
        g1[7] = 0;
        __builtin_amdgcn_tensor_load_to_lds(g0, g1, g2, g3, g4, 0);
        __builtin_amdgcn_s_wait_tensorcnt(0);
    }
    for (int i = tid; i < 512; i += 256) bias1[i] = b1[i];
    for (int i = tid; i < 128; i += 256) bias2[i] = b2[i];
    __syncthreads();

    // ---- GEMM1: K=256, nt = w + 8j, B panel in registers ------------------
    for (int j = 0; j < 4; ++j) {
        int nt = w + 8 * j;
        if (j < 3)
            __builtin_prefetch(&w1[(((size_t)(w + 8 * (j + 1))) * 32 + lane) * 16], 0, 1);
        BFrag bf[8];
#pragma unroll
        for (int kt = 0; kt < 8; ++kt) {
            const v4u* bp = (const v4u*)&w1[(((size_t)kt * 32 + nt) * 32 + lane) * 16];
            bf[kt].q[0] = bp[0];
            bf[kt].q[1] = bp[1];
        }
        for (int mt = 0; mt < 4; ++mt) {
            const int arow = (mt * 16 + rid) * USER_XS_S;
            v8f c = {};
#pragma unroll
            for (int kt = 0; kt < 8; ++kt) {
                int k0 = kt * 32 + khalf * 8;
                BFrag af;
                af.q[0] = *(const v4u*)&Xs[arow + k0];
                af.q[1] = *(const v4u*)&Xs[arow + k0 + 16];
                c = wmma_bf16(af.v, bf[kt].v, c);
            }
            int col = nt * 16 + rid;
            float bv = bias1[col];
            int r0 = mt * 16 + khalf * 8;
#pragma unroll
            for (int v = 0; v < 8; ++v) {
                float x = c[v] + bv;
                x = x > 0.f ? x : 0.f;
                Hs[(r0 + v) * HS_S + col] = f2bf(x);
            }
        }
    }
    __syncthreads();

    // ---- GEMM2: wave w owns nt = w ----------------------------------------
    {
        int nt = w;
        BFrag bf2[16];
#pragma unroll
        for (int kt = 0; kt < 16; ++kt) {
            const v4u* bp = (const v4u*)&w2[(((size_t)kt * 8 + nt) * 32 + lane) * 16];
            bf2[kt].q[0] = bp[0];
            bf2[kt].q[1] = bp[1];
        }
        for (int mt = 0; mt < 4; ++mt) {
            const int arow = (mt * 16 + rid) * HS_S;
            v8f c = {};
#pragma unroll
            for (int kt = 0; kt < 16; ++kt) {
                int k0 = kt * 32 + khalf * 8;
                BFrag af;
                af.q[0] = *(const v4u*)&Hs[arow + k0];
                af.q[1] = *(const v4u*)&Hs[arow + k0 + 16];
                c = wmma_bf16(af.v, bf2[kt].v, c);
            }
            int col = nt * 16 + rid;
            float bv = bias2[col];
            int r0 = mt * 16 + khalf * 8;
#pragma unroll
            for (int v = 0; v < 8; ++v)
                Outs[(r0 + v) * OUT_S + col] = c[v] + bv;
        }
    }
    __syncthreads();

    // ---- L2 norm + store ---------------------------------------------------
    int r = tid >> 2, part = tid & 3;
    float s = 0.f;
#pragma unroll
    for (int i = 0; i < 32; ++i) {
        float x = Outs[r * OUT_S + part * 32 + i];
        s += x * x;
    }
    red[tid] = s;
    __syncthreads();
    if (part == 0) {
        float t = red[tid] + red[tid + 1] + red[tid + 2] + red[tid + 3];
        rowscale[r] = 1.f / fmaxf(sqrtf(t), 1e-12f);
    }
    __syncthreads();
    int gr = base_row + r;
    float sc = rowscale[r];
#pragma unroll
    for (int i = 0; i < 32; ++i)
        out[(size_t)gr * TOWER_DIM + part * 32 + i] =
            Outs[r * OUT_S + part * 32 + i] * sc;
}

// ---------------------------------------------------------------------------
extern "C" void kernel_launch(void* const* d_in, const int* in_sizes, int n_in,
                              void* d_out, int out_size, void* d_ws, size_t ws_size,
                              hipStream_t stream) {
    const float* text     = (const float*)d_in[0];
    const float* gcn_item = (const float*)d_in[1];
    const float* gcn_user = (const float*)d_in[2];
    const int*   hist     = (const int*)d_in[3];
    const float* W1i = (const float*)d_in[4];
    const float* b1i = (const float*)d_in[5];
    const float* W2i = (const float*)d_in[6];
    const float* b2i = (const float*)d_in[7];
    const float* W1u = (const float*)d_in[8];
    const float* b1u = (const float*)d_in[9];
    const float* W2u = (const float*)d_in[10];
    const float* b2u = (const float*)d_in[11];
    float* out = (float*)d_out;

    char* ws = (char*)d_ws;
    size_t off = 0;
    float* item_emb = (float*)(ws + off);                off += (size_t)N_ITEMS * TOWER_DIM * 4;
    unsigned short* fused = (unsigned short*)(ws + off); off += (size_t)N_USERS * 256 * 2;
    unsigned short* w1i_b = (unsigned short*)(ws + off); off += (size_t)384 * 512 * 2;
    unsigned short* w2i_b = (unsigned short*)(ws + off); off += (size_t)512 * 128 * 2;
    unsigned short* w1u_b = (unsigned short*)(ws + off); off += (size_t)256 * 512 * 2;
    unsigned short* w2u_b = (unsigned short*)(ws + off); off += (size_t)512 * 128 * 2;

    (void)hipFuncSetAttribute(reinterpret_cast<const void*>(item_tower_kernel),
                              hipFuncAttributeMaxDynamicSharedMemorySize, ITEM_SMEM);
    (void)hipFuncSetAttribute(reinterpret_cast<const void*>(user_tower_kernel),
                              hipFuncAttributeMaxDynamicSharedMemorySize, USER_SMEM);

    convert_weights_kernel<<<(12 * 32 * 32 + 255) / 256, 256, 0, stream>>>(W1i, w1i_b, 384, 512);
    convert_weights_kernel<<<(16 *  8 * 32 + 255) / 256, 256, 0, stream>>>(W2i, w2i_b, 512, 128);
    convert_weights_kernel<<<( 8 * 32 * 32 + 255) / 256, 256, 0, stream>>>(W1u, w1u_b, 256, 512);
    convert_weights_kernel<<<(16 *  8 * 32 + 255) / 256, 256, 0, stream>>>(W2u, w2u_b, 512, 128);

    // item tower: ceil(100000/64) = 1563 blocks
    item_tower_kernel<<<1563, 256, ITEM_SMEM, stream>>>(
        text, gcn_item, w1i_b, b1i, w2i_b, b2i, item_emb);

    pool_kernel<<<N_USERS / 8, 256, 0, stream>>>(item_emb, hist, gcn_user, fused);

    user_tower_kernel<<<N_USERS / 64, 256, USER_SMEM, stream>>>(
        fused, w1u_b, b1u, w2u_b, b2u, out);
}
